// GenerationModule_59150289601203
// MI455X (gfx1250) — matrix operand order repaired and checked by reading, not compiled, not run on hardware
//
#include <hip/hip_runtime.h>
#include <hip/hip_bf16.h>

typedef __bf16 bf16;
typedef __attribute__((ext_vector_type(16))) __bf16 v16bf;
typedef __attribute__((ext_vector_type(8)))  __bf16 v8bf;
typedef __attribute__((ext_vector_type(8)))  float  v8f;

#define LRELU(v) ((v) > 0.0f ? (v) : 0.2f * (v))

// ---------------------------------------------------------------------------
// Small elementwise / GEMV-style kernels (style pipeline: tiny, VALU is fine)
// ---------------------------------------------------------------------------
__global__ void k_vec_add(const float* __restrict__ a, const float* __restrict__ b,
                          float* __restrict__ o, int n) {
  int i = blockIdx.x * blockDim.x + threadIdx.x;
  if (i < n) o[i] = a[i] + b[i];
}

// out[b*ostride + j] = act( dot(in[b*K..], w[j*K..]) + bias[j] )
// mode: 0 = none, 1 = leaky_relu(0.2), 2 = +1.0
__global__ void k_linear(const float* __restrict__ in, const float* __restrict__ w,
                         const float* __restrict__ bias, float* __restrict__ out,
                         int B, int K, int O, int ostride, int mode) {
  int idx = blockIdx.x * blockDim.x + threadIdx.x;
  if (idx >= B * O) return;
  int b = idx / O, j = idx - b * O;
  const float* ip = in + (size_t)b * K;
  const float* wp = w + (size_t)j * K;
  float acc = 0.f;
  for (int k = 0; k < K; ++k) acc = fmaf(ip[k], wp[k], acc);
  acc += bias[j];
  if (mode == 1) acc = LRELU(acc);
  else if (mode == 2) acc += 1.0f;
  out[(size_t)b * ostride + j] = acc;
}

// wsq[o*Cin+i] = sum_t w[o][i][t]^2
__global__ void k_wsq(const float* __restrict__ w, float* __restrict__ wsq,
                      int Cout, int Cin, int taps) {
  int idx = blockIdx.x * blockDim.x + threadIdx.x;
  if (idx >= Cout * Cin) return;
  const float* p = w + (size_t)idx * taps;
  float a = 0.f;
  for (int t = 0; t < taps; ++t) a = fmaf(p[t], p[t], a);
  wsq[idx] = a;
}

// dem[b*512+o] = rsqrt( sum_i wsq[o][i] * s[b][i]^2 + eps )
__global__ void k_demod(const float* __restrict__ wsq, const float* __restrict__ s,
                        float* __restrict__ dem, int B, int Cout, int Cin) {
  int idx = blockIdx.x * blockDim.x + threadIdx.x;
  if (idx >= B * Cout) return;
  int b = idx / Cout, o = idx - b * Cout;
  const float* wr = wsq + (size_t)o * Cin;
  const float* sp = s + (size_t)b * 512;
  float a = 0.f;
  for (int i = 0; i < Cin; ++i) { float si = sp[i]; a = fmaf(wr[i] * si, si, a); }
  dem[(size_t)b * 512 + o] = rsqrtf(a + 1e-8f);
}

// Pack weights f32 [Cout][Cin][ks][ks] -> bf16 [Cout16][Cin*KSLOT].
// Row-padded slot layout: slot = dy*(1<<rshift) + dx; zero where dx>=ks or dy>=ks.
__global__ void k_wprep(const float* __restrict__ w, bf16* __restrict__ wb,
                        int Cout, int Cout16, int Cin, int ks, int kslot,
                        int kshift, int rshift, int Kp) {
  int idx = blockIdx.x * blockDim.x + threadIdx.x;
  if (idx >= Cout16 * Kp) return;
  int o = idx / Kp;
  int k = idx - o * Kp;
  int cin = k >> kshift;
  int slot = k & (kslot - 1);
  int dy = slot >> rshift;
  int dx = slot & ((1 << rshift) - 1);
  float v = 0.f;
  if (o < Cout && dy < ks && dx < ks)
    v = w[((size_t)o * Cin + cin) * (ks * ks) + dy * ks + dx];
  wb[idx] = (bf16)v;
}

// xp0[b][c][1+y][1+x] = upsample2(const)[y][x] * s0[b][c]   (B=8,C=512,H=8,Hp=10)
__global__ void k_pad_const(const float* __restrict__ cst, const float* __restrict__ s,
                            bf16* __restrict__ xp) {
  int idx = blockIdx.x * blockDim.x + threadIdx.x;
  if (idx >= 8 * 512 * 64) return;
  int x = idx & 7, y = (idx >> 3) & 7, c = (idx >> 6) & 511, b = idx >> 15;
  float v = cst[c * 16 + (y >> 1) * 4 + (x >> 1)] * s[(size_t)b * 512 + c];
  xp[(((size_t)b * 512 + c) * 10 + 1 + y) * 10 + 1 + x] = (bf16)v;
}

// ---------------------------------------------------------------------------
// Implicit-GEMM conv, WMMA bf16 (16x16x32), fused modulate/demod/lrelu/upsample
// Template: kernel size, padded width, Cin, M-tiles per wave, epilogue mode.
// One B (im2col) fragment feeds MT wmmas -> scattered-load cost amortized.
// ---------------------------------------------------------------------------
struct ConvArgs {
  const bf16* xp;      // [B][Cin][WP][WP] padded, pre-scaled bf16 input
  const bf16* wb;      // [Cout16][Kp] packed bf16 weights (row-padded slots)
  const float* demod;  // [B][512] (mode 0)
  const float* bias;   // [Cout]
  const float* snext;  // [B][512] modulation scale for next layer input (or null)
  bf16* xpn;           // next layer's padded input (mode 0)
  float* out;          // final image (mode 1)
  int Cout;
  int Hpn, Wpn, padn, up;
};

__device__ __forceinline__ v8f wmma_bf16(v16bf av, v16bf bv, v8f acc) {
  return __builtin_amdgcn_wmma_f32_16x16x32_bf16(false, av, false, bv,
                                                 (short)0, acc, false, false);
}

__device__ __forceinline__ v16bf cat16(v8bf l, v8bf h) {
  return __builtin_shufflevector(l, h, 0, 1, 2, 3, 4, 5, 6, 7,
                                 8, 9, 10, 11, 12, 13, 14, 15);
}

template <int KS, int WP, int CIN, int MT, int MODE>
__global__ __launch_bounds__(256) void k_conv(ConvArgs a) {
  constexpr int PAD   = KS / 2;
  constexpr int H     = WP - 2 * PAD;     // output spatial (== W)
  constexpr int HW    = H * H;
  constexpr int HPWP  = WP * WP;
  constexpr int KSLOT = (KS == 3) ? 16 : 64;
  constexpr int Kp    = CIN * KSLOT;

  const int lane  = threadIdx.x & 31;
  const int wave  = threadIdx.x >> 5;
  const int hi    = lane >> 4;            // half-wave select
  const int lo    = lane & 15;
  const int b     = blockIdx.z;
  const int p0    = (blockIdx.x * 8 + wave) * 16;
  if (p0 >= HW) return;                   // wave-uniform: EXEC all-ones below
  const int p  = p0 + lo;
  const bool pv = (p < HW);
  const int pp = pv ? p : 0;
  const int py = pp / H;                  // H is a power-of-two constant
  const int px = pp - py * H;

  // A rows: lane lo -> M row; K = chunk*32 + hi*16 + {0..15}
  const bf16* __restrict__ aptr =
      a.wb + (size_t)(blockIdx.y * (MT * 16) + lo) * Kp + (hi << 4);
  // B: per-lane pixel base; 3x3: channel = 2*kt + hi; 7x7: rows 2*hi within ch.
  const bf16* __restrict__ xptr =
      a.xp + ((size_t)b * CIN + (KS == 3 ? hi : 0)) * HPWP + (py * WP + px) +
      (KS == 7 ? 2 * hi * WP : 0);

  v8f acc[MT];
#pragma unroll
  for (int m = 0; m < MT; ++m) acc[m] = (v8f){};

  if (KS == 3) {
    constexpr int nK = Kp >> 5;
    for (int kt = 0; kt < nK; ++kt) {
      __builtin_prefetch(aptr + 256, 0, 1);
      v16bf bv;
#pragma unroll
      for (int j = 0; j < 12; ++j)        // slots dy*4+dx; dx==3 is dead (w=0)
        bv[j] = xptr[(j >> 2) * WP + (j & 3)];
#pragma unroll
      for (int j = 12; j < 16; ++j) bv[j] = (bf16)0.0f;
#pragma unroll
      for (int m = 0; m < MT; ++m) {
        const v8bf* ap = (const v8bf*)(aptr + (size_t)m * 16 * Kp);
        acc[m] = wmma_bf16(cat16(ap[0], ap[1]), bv, acc[m]);
      }
      aptr += 32;
      xptr += 2 * HPWP;
    }
  } else {
    // KS == 7, KSLOT 64: channel c = q; slots = (kt&1)*32 + hi*16 + j,
    // mapped to rows dy = slot>>3, cols dx = slot&7 (dx==7 / dy==7 dead).
    constexpr int nQ = Kp >> 6;
    for (int q = 0; q < nQ; ++q) {
      __builtin_prefetch(aptr + 512, 0, 1);
      v16bf bv0, bv1;
#pragma unroll
      for (int j = 0; j < 16; ++j)
        bv0[j] = xptr[(j >> 3) * WP + (j & 7)];
#pragma unroll
      for (int j = 0; j < 16; ++j)
        bv1[j] = xptr[(4 + (j >> 3)) * WP + (j & 7)];
#pragma unroll
      for (int m = 0; m < MT; ++m) {
        const v8bf* ap = (const v8bf*)(aptr + (size_t)m * 16 * Kp);
        acc[m] = wmma_bf16(cat16(ap[0], ap[1]), bv0, acc[m]);
        acc[m] = wmma_bf16(cat16(ap[4], ap[5]), bv1, acc[m]);
      }
      aptr += 64;
      xptr += HPWP;
    }
  }

  // ---- epilogue: C/D layout lane L: N = L%16, M = (L/16)*8 + r
#pragma unroll
  for (int m = 0; m < MT; ++m) {
#pragma unroll
    for (int r = 0; r < 8; ++r) {
      const int o = (blockIdx.y * MT + m) * 16 + hi * 8 + r;
      if (!pv || o >= a.Cout) continue;
      if (MODE == 0) {
        float v = acc[m][r] * a.demod[(size_t)b * 512 + o] + a.bias[o];
        v = LRELU(v);
        if (a.snext) v *= a.snext[(size_t)b * 512 + o];
        bf16 q = (bf16)v;
        if (a.up) {  // 2x nearest upsample into next padded input
          const int y2 = 2 * py + a.padn, x2 = 2 * px + a.padn;
          bf16* d = a.xpn + (((size_t)b * a.Cout + o) * a.Hpn + y2) * a.Wpn + x2;
          d[0] = q; d[1] = q; d[a.Wpn] = q; d[a.Wpn + 1] = q;
        } else {
          bf16* d = a.xpn +
                    (((size_t)b * a.Cout + o) * a.Hpn + py + a.padn) * a.Wpn +
                    (px + a.padn);
          d[0] = q;
        }
      } else {  // final RGB: sigmoid -> f32 output
        float v = acc[m][r] + a.bias[o];
        v = 1.0f / (1.0f + __expf(-v));
        a.out[((size_t)b * a.Cout + o) * HW + p] = v;
      }
    }
  }
}

// ---------------------------------------------------------------------------
extern "C" void kernel_launch(void* const* d_in, const int* in_sizes, int n_in,
                              void* d_out, int out_size, void* d_ws, size_t ws_size,
                              hipStream_t stream) {
  (void)in_sizes; (void)out_size; (void)ws_size;
  if (n_in < 53) return;
  static const int CH[7] = {512, 512, 512, 512, 256, 128, 64};
  const float* w_id = (const float*)d_in[0];
  const float* w_m  = (const float*)d_in[1];
  const float* cst  = (const float*)d_in[2];
  const float *mw1[6], *mb1[6], *mw2[6], *mb2[6];
  const float *modw[6], *modmw[6], *modmb[6], *modb[6];
  for (int i = 0; i < 6; ++i) mw1[i]   = (const float*)d_in[3 + i];
  for (int i = 0; i < 6; ++i) mb1[i]   = (const float*)d_in[9 + i];
  for (int i = 0; i < 6; ++i) mw2[i]   = (const float*)d_in[15 + i];
  for (int i = 0; i < 6; ++i) mb2[i]   = (const float*)d_in[21 + i];
  for (int i = 0; i < 6; ++i) modw[i]  = (const float*)d_in[27 + i];
  for (int i = 0; i < 6; ++i) modmw[i] = (const float*)d_in[33 + i];
  for (int i = 0; i < 6; ++i) modmb[i] = (const float*)d_in[39 + i];
  for (int i = 0; i < 6; ++i) modb[i]  = (const float*)d_in[45 + i];
  const float* rgbw = (const float*)d_in[51];
  const float* rgbb = (const float*)d_in[52];

  // ---- workspace carve (~216 MB + slack) ----
  char* base = (char*)d_ws;
  size_t off = 0;
  auto carve = [&](size_t bytes) -> void* {
    void* p = base + off;
    off = (off + bytes + 255) & ~(size_t)255;
    return p;
  };
  float* f    = (float*)carve(4096 * 4);
  float* h    = (float*)carve(4096 * 4);
  float* sty  = (float*)carve(4096 * 4);
  float* sbuf = (float*)carve(6 * 4096 * 4);   // s per layer, stride 4096 floats
  float* dbuf = (float*)carve(6 * 4096 * 4);   // demod per layer
  float* wsq  = (float*)carve((size_t)512 * 512 * 4);
  bf16*  wb   = (bf16*)carve((size_t)512 * 8192 * 2);            // max packed weights
  const size_t szA = (size_t)8 * 64 * 262 * 262 * 2 + 65536;     // xp0/2/4/rgb + slack
  const size_t szB = (size_t)8 * 128 * 258 * 258 * 2 + 65536;    // xp1/3/5 + slack
  bf16* xpA = (bf16*)carve(szA);
  bf16* xpB = (bf16*)carve(szB);
  bf16* bufs[2] = {xpA, xpB};

  // ---- style pipeline (activation-independent, run first) ----
  k_vec_add<<<16, 256, 0, stream>>>(w_id, w_m, f, 4096);
  for (int i = 0; i < 6; ++i) {
    int Cin = CH[i], Cout = CH[i + 1];
    k_linear<<<16, 256, 0, stream>>>(f, mw1[i], mb1[i], h, 8, 512, 512, 512, 1);
    k_linear<<<16, 256, 0, stream>>>(h, mw2[i], mb2[i], sty, 8, 512, 512, 512, 0);
    k_linear<<<(8 * Cin + 255) / 256, 256, 0, stream>>>(
        sty, modmw[i], modmb[i], sbuf + (size_t)i * 4096, 8, 512, Cin, 512, 2);
    k_wsq<<<(Cout * Cin + 255) / 256, 256, 0, stream>>>(modw[i], wsq, Cout, Cin, 9);
    k_demod<<<(8 * Cout + 255) / 256, 256, 0, stream>>>(
        wsq, sbuf + (size_t)i * 4096, dbuf + (size_t)i * 4096, 8, Cout, Cin);
  }

  // ---- initial padded, modulated input (upsampled const, 8x8, pad 1) ----
  hipMemsetAsync(xpA, 0, (size_t)8 * 512 * 10 * 10 * 2, stream);
  k_pad_const<<<(8 * 512 * 64 + 255) / 256, 256, 0, stream>>>(cst, sbuf, xpA);

  // ---- conv layers 0..5 + RGB (i==6) ----
  static const int MTv[7] = {4, 4, 4, 4, 2, 2, 1};
  for (int i = 0; i <= 6; ++i) {
    const bool rgb   = (i == 6);
    const int Cin    = rgb ? 64 : CH[i];
    const int Cout   = rgb ? 3 : CH[i + 1];
    const int H      = rgb ? 256 : (8 << i);
    const int ks     = rgb ? 7 : 3;
    const int kslot  = rgb ? 64 : 16;
    const int kshift = rgb ? 6 : 4;
    const int rshift = rgb ? 3 : 2;
    const int Kp     = Cin * kslot;
    const int Cout16 = (Cout + 15) & ~15;
    const int HW     = H * H;

    // pack weights for this layer (row-padded slot layout)
    const float* wsrc = rgb ? rgbw : modw[i];
    const int tot = Cout16 * Kp;
    k_wprep<<<(tot + 255) / 256, 256, 0, stream>>>(wsrc, wb, Cout, Cout16, Cin,
                                                   ks, kslot, kshift, rshift, Kp);

    ConvArgs a = {};
    a.xp   = bufs[i & 1];
    a.wb   = wb;
    a.Cout = Cout;
    if (!rgb) {
      a.demod = dbuf + (size_t)i * 4096;
      a.bias  = modb[i];
      a.up    = (i < 5) ? 1 : 0;
      const int Hn   = a.up ? 2 * H : H;    // next-layer input spatial size
      const int padn = (i < 5) ? 1 : 3;     // next conv is 3x3, except RGB 7x7
      a.padn = padn; a.Hpn = Hn + 2 * padn; a.Wpn = Hn + 2 * padn;
      a.snext = (i < 5) ? (sbuf + (size_t)(i + 1) * 4096) : nullptr;
      a.xpn = bufs[(i + 1) & 1];
      // zero next padded buffer (borders stay 0; interior fully overwritten)
      hipMemsetAsync(a.xpn, 0, (size_t)8 * Cout * a.Hpn * a.Wpn * 2, stream);
    } else {
      a.bias = rgbb;
      a.out  = (float*)d_out;
    }
    dim3 grid((HW + 127) / 128, Cout16 / (16 * MTv[i]), 8);
    switch (i) {
      case 0: k_conv<3, 10, 512, 4, 0><<<grid, 256, 0, stream>>>(a); break;
      case 1: k_conv<3, 18, 512, 4, 0><<<grid, 256, 0, stream>>>(a); break;
      case 2: k_conv<3, 34, 512, 4, 0><<<grid, 256, 0, stream>>>(a); break;
      case 3: k_conv<3, 66, 512, 4, 0><<<grid, 256, 0, stream>>>(a); break;
      case 4: k_conv<3, 130, 256, 2, 0><<<grid, 256, 0, stream>>>(a); break;
      case 5: k_conv<3, 258, 128, 2, 0><<<grid, 256, 0, stream>>>(a); break;
      case 6: k_conv<7, 262, 64, 1, 1><<<grid, 256, 0, stream>>>(a); break;
    }
  }
}
